// ViT_33457795236327
// MI455X (gfx1250) — matrix operand order repaired and checked by reading, not compiled, not run on hardware
//
#include <hip/hip_runtime.h>
#include <hip/hip_bf16.h>
#include <math.h>

// ---------------- problem constants ----------------
#define B_     512
#define T_     98
#define F_     40
#define DIM_   192
#define DEPTH_ 12
#define HEADS_ 3
#define DHEAD_ 64
#define MLP_   768
#define NCLS_  35
#define INNER_ 192          // HEADS*DHEAD
#define N_     99           // T+1 tokens
#define M_     (B_ * N_)    // 50688 rows -> exactly 3168 16-row tiles
#define NP_    112          // tokens padded to 7x16 tiles for attention
#define KP_    128          // attention K padded to multiple of 32

typedef unsigned short u16;
typedef __bf16 v16bf __attribute__((ext_vector_type(16)));
typedef float  v8f   __attribute__((ext_vector_type(8)));
typedef unsigned int v4u __attribute__((ext_vector_type(4)));

union Frag { v16bf v; v4u q[2]; };

__device__ __forceinline__ u16 f2bf(float f) {
  union { float f; unsigned int u; } c; c.f = f;
  unsigned int u = c.u;
  return (u16)((u + 0x7fffu + ((u >> 16) & 1u)) >> 16);   // RNE f32 -> bf16
}

// ---------------- epilogue store for one 16x16 tile ----------------
// EPI: 0 = Cf32 = acc*scale           (attention scores)
//      1 = Cbf16 = acc                (qkv, attn@v)
//      2 = Cf32 = acc + bias + resid  (pre-LN sum)
//      3 = Cbf16 = gelu(acc + bias)   (FF1)
template<int EPI>
__device__ __forceinline__ void epi_store(const v8f& acc, int mt, int nt, int r, int half,
                                          float* __restrict__ Cf, u16* __restrict__ Cb,
                                          const float* __restrict__ bias,
                                          const float* __restrict__ resid,
                                          int Mstore, int Nc, int ldc, long long co,
                                          float scale)
{
  const int col = nt * 16 + r;
  if (col >= Nc) return;                         // Nc is a multiple of 16 -> tile-uniform
  const int rowBase = mt * 16 + half * 8;        // C/D: VGPR i -> M = i + 8*half
#pragma unroll
  for (int i = 0; i < 8; ++i) {
    const int row = rowBase + i;
    if (row >= Mstore) continue;
    const long long idx = co + (long long)row * ldc + col;
    const float v = acc[i];
    if (EPI == 0) {
      Cf[idx] = v * scale;
    } else if (EPI == 1) {
      Cb[idx] = f2bf(v);
    } else if (EPI == 2) {
      Cf[idx] = v + bias[col] + resid[idx];
    } else {
      const float t = v + bias[col];
      Cb[idx] = f2bf(0.5f * t * (1.0f + erff(t * 0.70710678118654752f)));
    }
  }
}

// ---------------- generic bf16 WMMA GEMM, 2x2 register-tiled ----------------
// C[M x Nc] = A[M x K](bf16, row-major, lda) * Wt[Nc x K](bf16, row-major, ldw)
// Each wave computes a 32x32 output block (4 WMMA tiles) so every A/B fragment
// is reused twice: 8 b128 loads feed 4 v_wmma (2:1 vmem:wmma).
// Optional (b,h) batching via blockIdx.z with independent element strides.
template<int EPI>
__global__ __launch_bounds__(256)
void wmma_gemm(const u16* __restrict__ A, const u16* __restrict__ W,
               float* __restrict__ Cf, u16* __restrict__ Cb,
               const float* __restrict__ bias, const float* __restrict__ resid,
               int M, int Mstore, int Nc, int K,
               int lda, int ldw, int ldc, float scale, int H,
               long long sAb, long long sAh, long long sWb, long long sWh,
               long long sCb, long long sCh)
{
  const int lane = threadIdx.x & 31;
  const int wave = threadIdx.x >> 5;
  const int Mt = (M + 15) >> 4, Nt = (Nc + 15) >> 4;
  // 8 waves/block arranged 2(m) x 4(n); each wave owns 2x2 tiles (32x32 out).
  const int mt0 = (blockIdx.x * 2 + (wave >> 2)) * 2;
  const int nt0 = (blockIdx.y * 4 + (wave & 3)) * 2;
  if (mt0 >= Mt || nt0 >= Nt) return;            // wave-uniform: EXEC stays all-1
  const int mt1c = (mt0 + 1 < Mt) ? mt0 + 1 : Mt - 1;   // clamped for loads
  const int nt1c = (nt0 + 1 < Nt) ? nt0 + 1 : Nt - 1;

  long long ao = 0, wo = 0, co = 0;
  if (H > 0) {
    const int z = blockIdx.z;
    const int bb = z / H, hh = z - bb * H;
    ao = (long long)bb * sAb + (long long)hh * sAh;
    wo = (long long)bb * sWb + (long long)hh * sWh;
    co = (long long)bb * sCb + (long long)hh * sCh;
  }
  const int r    = lane & 15;
  const int half = lane >> 4;
  // A 16x32 bf16 layout (ISA 7.12.2): lane half 0 -> K 0..7 & 16..23, half 1 -> +8
  const u16* ap0 = A + ao + (long long)(mt0  * 16 + r) * lda + half * 8;
  const u16* ap1 = A + ao + (long long)(mt1c * 16 + r) * lda + half * 8;
  // B 32x16 layout (same as A): lane half 0 -> K 0..7 & 16..23, half 1 -> +8
  const u16* wp0 = W + wo + (long long)(nt0  * 16 + r) * ldw + half * 8;
  const u16* wp1 = W + wo + (long long)(nt1c * 16 + r) * ldw + half * 8;

  v8f a00 = {}, a01 = {}, a10 = {}, a11 = {};
  for (int k0 = 0; k0 < K; k0 += 32) {
    Frag fa0, fa1, fb0, fb1;
    fa0.q[0] = *(const v4u*)(ap0 + k0);
    fa0.q[1] = *(const v4u*)(ap0 + k0 + 16);
    fa1.q[0] = *(const v4u*)(ap1 + k0);
    fa1.q[1] = *(const v4u*)(ap1 + k0 + 16);
    fb0.q[0] = *(const v4u*)(wp0 + k0);
    fb0.q[1] = *(const v4u*)(wp0 + k0 + 16);
    fb1.q[0] = *(const v4u*)(wp1 + k0);
    fb1.q[1] = *(const v4u*)(wp1 + k0 + 16);
    a00 = __builtin_amdgcn_wmma_f32_16x16x32_bf16(false, fa0.v, false, fb0.v, (short)0, a00, false, false);
    a01 = __builtin_amdgcn_wmma_f32_16x16x32_bf16(false, fa0.v, false, fb1.v, (short)0, a01, false, false);
    a10 = __builtin_amdgcn_wmma_f32_16x16x32_bf16(false, fa1.v, false, fb0.v, (short)0, a10, false, false);
    a11 = __builtin_amdgcn_wmma_f32_16x16x32_bf16(false, fa1.v, false, fb1.v, (short)0, a11, false, false);
  }

  // Stores use UNCLAMPED tile coordinates: clamp-duplicated tiles fall outside
  // [0,Mstore) x [0,Nc) and are dropped by the guards.
  epi_store<EPI>(a00, mt0,     nt0,     r, half, Cf, Cb, bias, resid, Mstore, Nc, ldc, co, scale);
  epi_store<EPI>(a01, mt0,     nt0 + 1, r, half, Cf, Cb, bias, resid, Mstore, Nc, ldc, co, scale);
  epi_store<EPI>(a10, mt0 + 1, nt0,     r, half, Cf, Cb, bias, resid, Mstore, Nc, ldc, co, scale);
  epi_store<EPI>(a11, mt0 + 1, nt0 + 1, r, half, Cf, Cb, bias, resid, Mstore, Nc, ldc, co, scale);
}

// ---------------- weight transpose + f32->bf16 ----------------
// src: K x Nc (row-major f32), dst: Nc x K (row-major bf16)
__global__ void transpose_w(const float* __restrict__ src, u16* __restrict__ dst,
                            int K, int Nc) {
  int i = blockIdx.x * blockDim.x + threadIdx.x;
  if (i >= K * Nc) return;
  int n = i / K, k = i - n * K;
  dst[i] = f2bf(src[(long long)k * Nc + n]);
}

// ---------------- patch embed + cls + pos ----------------
__global__ void embed_k(const float* __restrict__ img, const float* __restrict__ lin_w,
                        const float* __restrict__ g_feat, const float* __restrict__ pos,
                        float* __restrict__ xf, u16* __restrict__ xb) {
  int i = blockIdx.x * blockDim.x + threadIdx.x;
  if (i >= B_ * N_ * DIM_) return;
  int d = i % DIM_; int tb = i / DIM_; int t = tb % N_; int b = tb / N_;
  float v;
  if (t == 0) {
    v = g_feat[d];
  } else {
    const float* ip = img + ((long long)b * T_ + (t - 1)) * F_;
    float s = 0.f;
#pragma unroll 8
    for (int f = 0; f < F_; ++f) s += ip[f] * lin_w[f * DIM_ + d];
    v = s;
  }
  v += pos[t * DIM_ + d];
  xf[i] = v; xb[i] = f2bf(v);
}

// ---------------- softmax (f32 in, zero-padded bf16 out) ----------------
// grid.x = B*H*NP rows; scores row stride NP_, attn row stride KP_
__global__ __launch_bounds__(128)
void softmax_k(const float* __restrict__ scores, u16* __restrict__ attn) {
  const int rowg = blockIdx.x;
  const int row  = rowg % NP_;
  const long long sbase = (long long)rowg * NP_;
  const long long abase = (long long)rowg * KP_;
  const int tid = threadIdx.x;
  __shared__ float red[128];
  float val = -1e30f;
  if (row < N_ && tid < N_) val = scores[sbase + tid];
  red[tid] = val; __syncthreads();
  for (int s = 64; s > 0; s >>= 1) {
    if (tid < s) red[tid] = fmaxf(red[tid], red[tid + s]);
    __syncthreads();
  }
  const float mx = red[0]; __syncthreads();
  float e = 0.f;
  if (row < N_ && tid < N_) e = __expf(val - mx);
  red[tid] = e; __syncthreads();
  for (int s = 64; s > 0; s >>= 1) {
    if (tid < s) red[tid] += red[tid + s];
    __syncthreads();
  }
  const float inv = (row < N_) ? 1.0f / red[0] : 0.f;
  attn[abase + tid] = f2bf(e * inv);   // cols >= N_ and rows >= N_ become exact zeros
}

// ---------------- V transpose: qkv bf16 -> Vt[b][h][d][tk] (tk padded 128) -----
__global__ void vt_k(const u16* __restrict__ qkvb, u16* __restrict__ vt) {
  int i = blockIdx.x * blockDim.x + threadIdx.x;
  if (i >= B_ * HEADS_ * DHEAD_ * KP_) return;
  int tk = i & (KP_ - 1); int rest = i >> 7;
  int d = rest & (DHEAD_ - 1); rest >>= 6;
  int h = rest % HEADS_; int b = rest / HEADS_;
  u16 v = 0;
  if (tk < N_) v = qkvb[((long long)(b * N_ + tk)) * (3 * INNER_) + 2 * INNER_ + h * DHEAD_ + d];
  vt[i] = v;
}

// ---------------- LayerNorm over DIM, writes f32 + bf16 ----------------
__global__ __launch_bounds__(192)
void ln_k(const float* __restrict__ src, const float* __restrict__ g,
          const float* __restrict__ bt, float* __restrict__ xf, u16* __restrict__ xb) {
  const long long base = (long long)blockIdx.x * DIM_;
  const int tid = threadIdx.x;
  const float v = src[base + tid];
  __shared__ float s1[192], s2[192], mv[2];
  s1[tid] = v; s2[tid] = v * v; __syncthreads();
  for (int s = 96; s >= 6; s >>= 1) {
    if (tid < s) { s1[tid] += s1[tid + s]; s2[tid] += s2[tid + s]; }
    __syncthreads();
  }
  if (tid == 0) {
    float a = 0.f, b = 0.f;
    for (int k = 0; k < 6; ++k) { a += s1[k]; b += s2[k]; }
    const float m = a / DIM_;
    mv[0] = m; mv[1] = rsqrtf(b / DIM_ - m * m + 1e-5f);
  }
  __syncthreads();
  const float o = (v - mv[0]) * mv[1] * g[tid] + bt[tid];
  xf[base + tid] = o; xb[base + tid] = f2bf(o);
}

// ---------------- classifier head: LN(x[:,0]) @ head_w + head_b ----------------
__global__ __launch_bounds__(192)
void head_k(const float* __restrict__ xf, const float* __restrict__ g,
            const float* __restrict__ bt, const float* __restrict__ hw,
            const float* __restrict__ hb, float* __restrict__ out) {
  const int b = blockIdx.x;
  const int tid = threadIdx.x;
  const long long base = (long long)b * N_ * DIM_;   // token 0 of batch b
  const float v = xf[base + tid];
  __shared__ float s1[192], s2[192], sn[192], mv[2];
  s1[tid] = v; s2[tid] = v * v; __syncthreads();
  for (int s = 96; s >= 6; s >>= 1) {
    if (tid < s) { s1[tid] += s1[tid + s]; s2[tid] += s2[tid + s]; }
    __syncthreads();
  }
  if (tid == 0) {
    float a = 0.f, c = 0.f;
    for (int k = 0; k < 6; ++k) { a += s1[k]; c += s2[k]; }
    const float m = a / DIM_;
    mv[0] = m; mv[1] = rsqrtf(c / DIM_ - m * m + 1e-5f);
  }
  __syncthreads();
  sn[tid] = (v - mv[0]) * mv[1] * g[tid] + bt[tid];
  __syncthreads();
  if (tid < NCLS_) {
    float acc = hb[tid];
    for (int d = 0; d < DIM_; ++d) acc += sn[d] * hw[d * NCLS_ + tid];
    out[b * NCLS_ + tid] = acc;
  }
}

// ---------------- host orchestration ----------------
static inline dim3 gemm_grid(int M, int Nc, int Z) {
  const int mt = (M + 15) / 16, nt = (Nc + 15) / 16;
  // block covers 4 m-tiles (2 waves x 2) and 8 n-tiles (4 waves x 2)
  return dim3((unsigned)((mt + 3) / 4), (unsigned)((nt + 7) / 8), (unsigned)Z);
}

extern "C" void kernel_launch(void* const* d_in, const int* in_sizes, int n_in,
                              void* d_out, int out_size, void* d_ws, size_t ws_size,
                              hipStream_t stream) {
  (void)in_sizes; (void)n_in; (void)out_size; (void)ws_size;
  const float* img       = (const float*)d_in[0];
  const float* lin_w     = (const float*)d_in[1];
  const float* g_feat    = (const float*)d_in[2];
  const float* pos_emb   = (const float*)d_in[3];
  const float* ln1_g     = (const float*)d_in[4];
  const float* ln1_b     = (const float*)d_in[5];
  const float* ln2_g     = (const float*)d_in[6];
  const float* ln2_b     = (const float*)d_in[7];
  const float* qkv_w     = (const float*)d_in[8];
  const float* out_w     = (const float*)d_in[9];
  const float* out_b     = (const float*)d_in[10];
  const float* ff1_w     = (const float*)d_in[11];
  const float* ff1_b     = (const float*)d_in[12];
  const float* ff2_w     = (const float*)d_in[13];
  const float* ff2_b     = (const float*)d_in[14];
  const float* head_ln_g = (const float*)d_in[15];
  const float* head_ln_b = (const float*)d_in[16];
  const float* head_w    = (const float*)d_in[17];
  const float* head_b    = (const float*)d_in[18];

  char* ws = (char*)d_ws;
  size_t off = 0;
  auto take = [&](size_t bytes) { size_t o = off; off += (bytes + 255) & ~(size_t)255; return o; };

  float* xf    = (float*)(ws + take((size_t)M_ * DIM_ * 4));
  u16*   xb    = (u16*)  (ws + take((size_t)M_ * DIM_ * 2));
  float* tmp   = (float*)(ws + take((size_t)M_ * DIM_ * 4));
  u16*   ob    = (u16*)  (ws + take((size_t)M_ * DIM_ * 2));
  u16*   qkvb  = (u16*)  (ws + take((size_t)(M_ + 32) * 3 * INNER_ * 2)); // +pad rows for tile overrun
  u16*   qkvWt = (u16*)  (ws + take((size_t)DEPTH_ * 3 * INNER_ * DIM_ * 2));
  u16*   outWt = (u16*)  (ws + take((size_t)DEPTH_ * DIM_ * INNER_ * 2));
  u16*   ff1Wt = (u16*)  (ws + take((size_t)DEPTH_ * MLP_ * DIM_ * 2));
  u16*   ff2Wt = (u16*)  (ws + take((size_t)DEPTH_ * DIM_ * MLP_ * 2));
  // attention region (aliased with MLP hidden buffer: disjoint lifetimes)
  const size_t region = off;
  const size_t sc_sz  = ((size_t)B_ * HEADS_ * NP_ * NP_ * 4 + 255) & ~(size_t)255;
  const size_t at_sz  = ((size_t)B_ * HEADS_ * NP_ * KP_ * 2 + 255) & ~(size_t)255;
  float* scores = (float*)(ws + region);
  u16*   attn   = (u16*)  (ws + region + sc_sz);
  u16*   vt     = (u16*)  (ws + region + sc_sz + at_sz);
  u16*   hb     = (u16*)  (ws + region);            // alias over scores/attn

  // ---- weight prep (bf16, transposed to [Nout][K]) ----
  for (int l = 0; l < DEPTH_; ++l) {
    transpose_w<<<(DIM_ * 3 * INNER_ + 255) / 256, 256, 0, stream>>>(
        qkv_w + (size_t)l * DIM_ * 3 * INNER_, qkvWt + (size_t)l * 3 * INNER_ * DIM_, DIM_, 3 * INNER_);
    transpose_w<<<(INNER_ * DIM_ + 255) / 256, 256, 0, stream>>>(
        out_w + (size_t)l * INNER_ * DIM_, outWt + (size_t)l * DIM_ * INNER_, INNER_, DIM_);
    transpose_w<<<(DIM_ * MLP_ + 255) / 256, 256, 0, stream>>>(
        ff1_w + (size_t)l * DIM_ * MLP_, ff1Wt + (size_t)l * MLP_ * DIM_, DIM_, MLP_);
    transpose_w<<<(MLP_ * DIM_ + 255) / 256, 256, 0, stream>>>(
        ff2_w + (size_t)l * MLP_ * DIM_, ff2Wt + (size_t)l * DIM_ * MLP_, MLP_, DIM_);
  }

  // ---- embed ----
  embed_k<<<(B_ * N_ * DIM_ + 255) / 256, 256, 0, stream>>>(img, lin_w, g_feat, pos_emb, xf, xb);

  // ---- transformer layers ----
  for (int l = 0; l < DEPTH_; ++l) {
    // qkv = x @ qkv_w  -> bf16 [M][576]
    wmma_gemm<1><<<gemm_grid(M_, 3 * INNER_, 1), 256, 0, stream>>>(
        xb, qkvWt + (size_t)l * 3 * INNER_ * DIM_, nullptr, qkvb, nullptr, nullptr,
        M_, M_, 3 * INNER_, DIM_, DIM_, DIM_, 3 * INNER_, 1.f, 0, 0, 0, 0, 0, 0, 0);

    // scores = (q @ k^T) * 1/8, batched over (b,h)
    wmma_gemm<0><<<gemm_grid(NP_, NP_, B_ * HEADS_), 256, 0, stream>>>(
        qkvb, qkvb + INNER_, scores, nullptr, nullptr, nullptr,
        NP_, NP_, NP_, DHEAD_, 3 * INNER_, 3 * INNER_, NP_, 0.125f, HEADS_,
        (long long)N_ * 3 * INNER_, DHEAD_,
        (long long)N_ * 3 * INNER_, DHEAD_,
        (long long)HEADS_ * NP_ * NP_, (long long)NP_ * NP_);

    softmax_k<<<B_ * HEADS_ * NP_, 128, 0, stream>>>(scores, attn);
    vt_k<<<(B_ * HEADS_ * DHEAD_ * KP_ + 255) / 256, 256, 0, stream>>>(qkvb, vt);

    // o = attn @ v  -> bf16 [M][192] (heads merged via per-head col offset)
    wmma_gemm<1><<<gemm_grid(NP_, DHEAD_, B_ * HEADS_), 256, 0, stream>>>(
        attn, vt, nullptr, ob, nullptr, nullptr,
        NP_, N_, DHEAD_, KP_, KP_, KP_, INNER_, 1.f, HEADS_,
        (long long)HEADS_ * NP_ * KP_, (long long)NP_ * KP_,
        (long long)HEADS_ * DHEAD_ * KP_, (long long)DHEAD_ * KP_,
        (long long)N_ * INNER_, DHEAD_);

    // tmp = o @ out_w + out_b + x
    wmma_gemm<2><<<gemm_grid(M_, DIM_, 1), 256, 0, stream>>>(
        ob, outWt + (size_t)l * DIM_ * INNER_, tmp, nullptr,
        out_b + (size_t)l * DIM_, xf,
        M_, M_, DIM_, INNER_, INNER_, INNER_, DIM_, 1.f, 0, 0, 0, 0, 0, 0, 0);

    ln_k<<<M_, 192, 0, stream>>>(tmp, ln1_g + l * DIM_, ln1_b + l * DIM_, xf, xb);

    // h = gelu(x @ ff1_w + ff1_b) -> bf16 [M][768]
    wmma_gemm<3><<<gemm_grid(M_, MLP_, 1), 256, 0, stream>>>(
        xb, ff1Wt + (size_t)l * MLP_ * DIM_, nullptr, hb,
        ff1_b + (size_t)l * MLP_, nullptr,
        M_, M_, MLP_, DIM_, DIM_, DIM_, MLP_, 1.f, 0, 0, 0, 0, 0, 0, 0);

    // tmp = h @ ff2_w + ff2_b + x
    wmma_gemm<2><<<gemm_grid(M_, DIM_, 1), 256, 0, stream>>>(
        hb, ff2Wt + (size_t)l * DIM_ * MLP_, tmp, nullptr,
        ff2_b + (size_t)l * DIM_, xf,
        M_, M_, DIM_, MLP_, MLP_, MLP_, DIM_, 1.f, 0, 0, 0, 0, 0, 0, 0);

    ln_k<<<M_, 192, 0, stream>>>(tmp, ln2_g + l * DIM_, ln2_b + l * DIM_, xf, xb);
  }

  // ---- head ----
  head_k<<<B_, 192, 0, stream>>>(xf, head_ln_g, head_ln_b, head_w, head_b, (float*)d_out);
}
